// student_38010460570240
// MI455X (gfx1250) — compile-verified, matrix operand order checked by reading, not run on hardware
//
#include <hip/hip_runtime.h>
#include <cstdint>

#define EMBED 768
#define HEADS 12
#define HEAD_DIM 64
#define TOKENS 4096
#define MLPDIM 3072

typedef __attribute__((ext_vector_type(8)))  float  v8f;
typedef __attribute__((ext_vector_type(16))) __bf16 v16bf;

struct __attribute__((aligned(16))) u128 { unsigned int x, y, z, w; };
union ABFrag { v16bf v; u128 u[2]; };

#define WMMA_BF16(a, b, c) \
  __builtin_amdgcn_wmma_f32_16x16x32_bf16(false, (a), false, (b), (short)0, (c), false, false)

#if __has_builtin(__builtin_amdgcn_tensor_load_to_lds)
#define USE_TDM 1
#else
#define USE_TDM 0
#endif

// ---------------------------------------------------------------------------
// Generic bf16 GEMM: C[M,N] = act(A[M,K] * W[N,K]^T + bias + addend)
// Tile: 128 (M) x 64 (N), K-step 32.  8 waves, each owns a 16-row strip.
// Double-buffered LDS; A staged via pipelined global loads, B via the
// Tensor Data Mover.  One barrier per K-step; WMMAs issue back-to-back.
// ---------------------------------------------------------------------------
__global__ __launch_bounds__(256) void gemm_bf16_kernel(
    const __bf16* __restrict__ A, const __bf16* __restrict__ W,
    const float* __restrict__ bias, const float* __restrict__ addend,
    float* __restrict__ outF, __bf16* __restrict__ outB,
    int M, int N, int K, int actGelu)
{
  (void)M;
  __shared__ __align__(16) __bf16 At[2][128 * 32];
  __shared__ __align__(16) __bf16 Bt[2][64 * 32];

  const int tid  = threadIdx.x;
  const int lane = tid & 31;
  const int wave = tid >> 5;
  const int mBase = blockIdx.y * 128;
  const int nBase = blockIdx.x * 64;
  const int r16   = lane & 15;
  const int hiOff = (lane >> 4) ? 8 : 0;

  // This thread's two A-tile staging chunks (128 rows x 32 halves).
  const int rowA0 = tid >> 2,           segA0 = tid & 3;
  const int rowA1 = (tid + 256) >> 2,   segA1 = (tid + 256) & 3;

  auto loadA = [&](int k0, u128& a0, u128& a1) {
    a0 = *(const u128*)(A + (size_t)(mBase + rowA0) * K + k0 + segA0 * 8);
    a1 = *(const u128*)(A + (size_t)(mBase + rowA1) * K + k0 + segA1 * 8);
  };
  auto storeA = [&](int buf, const u128& a0, const u128& a1) {
    *(u128*)(At[buf] + rowA0 * 32 + segA0 * 8) = a0;
    *(u128*)(At[buf] + rowA1 * 32 + segA1 * 8) = a1;
  };
  auto stageB = [&](int k0, int buf) {
#if USE_TDM
    if (wave == 0) {
      typedef unsigned int u32x4 __attribute__((ext_vector_type(4)));
      typedef int          i32x8 __attribute__((ext_vector_type(8)));
      typedef int          i32x4 __attribute__((ext_vector_type(4)));
      unsigned long long ga = (unsigned long long)(uintptr_t)(W + (size_t)nBase * K + k0);
      unsigned int ldsa = (unsigned int)(uintptr_t)(&Bt[buf][0]);
      u32x4 g0;
      g0[0] = 1u;                                   // count=1, user descriptor
      g0[1] = ldsa;                                 // lds_addr
      g0[2] = (unsigned int)ga;                     // global_addr lo
      g0[3] = (unsigned int)((ga >> 32) & 0x1FFFFFFu) | (2u << 30);  // addr hi | type=2
      i32x8 g1;
      g1[0] = (1 << 16);                            // workgroup_mask=0, data_size=2B
      g1[1] = (int)(((unsigned)K & 0xFFFFu) << 16); // tensor_dim0[15:0]
      g1[2] = (int)(((unsigned)K >> 16) | (((unsigned)N & 0xFFFFu) << 16)); // dim0 hi|dim1 lo
      g1[3] = (int)(((unsigned)N >> 16) | (32u << 16)); // dim1 hi | tile_dim0=32
      g1[4] = 64;                                   // tile_dim1=64, tile_dim2=0
      g1[5] = K;                                    // tensor_dim0_stride lo32
      g1[6] = 0;
      g1[7] = 0;
      i32x4 z4 = {0, 0, 0, 0};
      i32x8 z8 = {0, 0, 0, 0, 0, 0, 0, 0};
      __builtin_amdgcn_tensor_load_to_lds(g0, g1, z4, z4, z8, 0);
    }
#else
    if (tid < 256) {
      int row = tid >> 2, seg = tid & 3;
      *(u128*)(Bt[buf] + row * 32 + seg * 8) =
          *(const u128*)(W + (size_t)(nBase + row) * K + k0 + seg * 8);
    }
#endif
  };

  v8f acc[4] = {};

  // Prologue: stage tile 0.
  {
    u128 a0, a1;
    loadA(0, a0, a1);
    storeA(0, a0, a1);
    stageB(0, 0);
  }
#if USE_TDM
  __builtin_amdgcn_s_wait_tensorcnt(0);
#endif
  __syncthreads();

  for (int k0 = 0; k0 < K; k0 += 32) {
    const int cur = (k0 >> 5) & 1, nxt = cur ^ 1;
    const bool more = (k0 + 32) < K;

    // Kick off next tile: global A loads stay in flight during compute,
    // TDM streams the next weight tile into the alternate LDS buffer.
    u128 a0, a1;
    if (more) {
      loadA(k0 + 32, a0, a1);
      stageB(k0 + 32, nxt);
      __builtin_prefetch(A + (size_t)(mBase + rowA0) * K + k0 + 64, 0, 1);
    }

    // Compute on current buffers: batch all fragment loads, then 4 WMMAs.
    ABFrag af, bf0, bf1, bf2, bf3;
    af.u[0]  = *(const u128*)(At[cur] + (wave * 16 + r16) * 32 + hiOff);
    af.u[1]  = *(const u128*)(At[cur] + (wave * 16 + r16) * 32 + hiOff + 16);
    bf0.u[0] = *(const u128*)(Bt[cur] + (r16)      * 32 + hiOff);
    bf0.u[1] = *(const u128*)(Bt[cur] + (r16)      * 32 + hiOff + 16);
    bf1.u[0] = *(const u128*)(Bt[cur] + (16 + r16) * 32 + hiOff);
    bf1.u[1] = *(const u128*)(Bt[cur] + (16 + r16) * 32 + hiOff + 16);
    bf2.u[0] = *(const u128*)(Bt[cur] + (32 + r16) * 32 + hiOff);
    bf2.u[1] = *(const u128*)(Bt[cur] + (32 + r16) * 32 + hiOff + 16);
    bf3.u[0] = *(const u128*)(Bt[cur] + (48 + r16) * 32 + hiOff);
    bf3.u[1] = *(const u128*)(Bt[cur] + (48 + r16) * 32 + hiOff + 16);
    acc[0] = WMMA_BF16(af.v, bf0.v, acc[0]);
    acc[1] = WMMA_BF16(af.v, bf1.v, acc[1]);
    acc[2] = WMMA_BF16(af.v, bf2.v, acc[2]);
    acc[3] = WMMA_BF16(af.v, bf3.v, acc[3]);

    if (more) {
      storeA(nxt, a0, a1);
#if USE_TDM
      __builtin_amdgcn_s_wait_tensorcnt(0);
#endif
    }
    __syncthreads();
  }

  // Epilogue: lane holds column (t*16 + r16), rows hiOff..hiOff+7 of its strip.
#pragma unroll
  for (int t = 0; t < 4; ++t) {
    int n = nBase + t * 16 + r16;
    float bv = bias ? bias[n] : 0.f;
#pragma unroll
    for (int i = 0; i < 8; ++i) {
      int m = mBase + wave * 16 + hiOff + i;
      float v = acc[t][i] + bv;
      if (addend) v += addend[(size_t)m * N + n];
      if (actGelu) v = 0.5f * v * (1.f + erff(v * 0.70710678f));
      if (outF) outF[(size_t)m * N + n] = v;
      if (outB) outB[(size_t)m * N + n] = (__bf16)v;
    }
  }
}

// ---------------------------------------------------------------------------
// Flash attention with decomposed relative-position bias.
// Block = 4 waves; each wave owns a 16-row Q tile; K/V tiles of 32 in LDS.
// winPerSide == 0 => global attention (size = 64); else windowed (size = 14).
// ---------------------------------------------------------------------------
__global__ __launch_bounds__(128) void attn_kernel(
    const __bf16* __restrict__ qkv, __bf16* __restrict__ out,
    const float* __restrict__ relH, const float* __restrict__ relW,
    int N, int size, int winPerSide)
{
  __shared__ __align__(16) __bf16 Kt[32 * 64];
  __shared__ __align__(16) __bf16 Vt[64 * 32];   // transposed: [hd][key]
  __shared__ __align__(16) __bf16 Pl[4][16 * 32];
  __shared__ float RH[4][16 * 64];
  __shared__ float RW[4][16 * 64];

  const int tid = threadIdx.x, lane = tid & 31, wave = tid >> 5;
  const int head = blockIdx.x % HEADS;
  const int win  = blockIdx.x / HEADS;
  const int qbase = blockIdx.y * 64 + wave * 16;
  const int r16 = lane & 15;
  const int hiOff = (lane >> 4) ? 8 : 0;
  const float scale = 0.125f;  // 64^-0.5

  auto mapTok = [&](int l, int& tok, int& qh, int& qw) -> bool {
    if (l >= N) return false;
    if (winPerSide == 0) { tok = l; qh = l >> 6; qw = l & 63; return true; }
    int r = l / size, c = l - r * size;
    qh = r; qw = c;
    int gy = (win / winPerSide) * size + r;
    int gx = (win % winPerSide) * size + c;
    if (gy >= 64 || gx >= 64) return false;
    tok = gy * 64 + gx;
    return true;
  };

  // Q fragments (hd chunks 0..31 and 32..63), one row per lane.
  ABFrag aq0 = {}, aq1 = {};
  {
    int l = qbase + r16, tok, a, b;
    if (mapTok(l, tok, a, b)) {
      const __bf16* qp = qkv + (size_t)tok * 2304 + head * 64;
      aq0.u[0] = *(const u128*)(qp + hiOff);
      aq0.u[1] = *(const u128*)(qp + hiOff + 16);
      aq1.u[0] = *(const u128*)(qp + 32 + hiOff);
      aq1.u[1] = *(const u128*)(qp + 32 + hiOff + 16);
    }
  }
  int rowTok[8]; bool rowValid[8];
#pragma unroll
  for (int i = 0; i < 8; ++i) {
    int t = 0, a, b;
    rowValid[i] = mapTok(qbase + hiOff + i, t, a, b);
    rowTok[i] = t;
  }

  // Precompute rel-pos tables: RH[row][kh] = q_row . Rh[qh,kh,:], same for RW.
  {
    int m = r16, tok, qh, qw;
    bool ok = mapTok(qbase + m, tok, qh, qw);
    for (int kh = lane >> 4; kh < size; kh += 2) {
      float sh = 0.f, sw = 0.f;
      if (ok) {
        const __bf16* qp = qkv + (size_t)tok * 2304 + head * 64;
        const float* rh = relH + (size_t)(qh - kh + size - 1) * HEAD_DIM;
        const float* rw = relW + (size_t)(qw - kh + size - 1) * HEAD_DIM;
        for (int c = 0; c < HEAD_DIM; ++c) {
          float q = (float)qp[c];
          sh += q * rh[c];
          sw += q * rw[c];
        }
      }
      RH[wave][m * 64 + kh] = sh;
      RW[wave][m * 64 + kh] = sw;
    }
  }
  __syncthreads();

  float mrow[8], lrow[8];
  v8f o[4] = {};
#pragma unroll
  for (int i = 0; i < 8; ++i) { mrow[i] = -1e30f; lrow[i] = 0.f; }

  const int nkt = (N + 31) / 32;
  for (int kt = 0; kt < nkt; ++kt) {
    __syncthreads();
    // Stage K (row-major) and V (transposed), zero-filling invalid keys.
#pragma unroll
    for (int it = 0; it < 2; ++it) {
      int chunk = tid + it * 128;       // 32 keys * 8 segments
      int kr = chunk >> 3, seg = chunk & 7;
      int l = kt * 32 + kr, tok, a, b;
      u128 kv = {0, 0, 0, 0}, vv = {0, 0, 0, 0};
      if (mapTok(l, tok, a, b)) {
        const __bf16* kp = qkv + (size_t)tok * 2304 + 768 + head * 64;
        const __bf16* vp = qkv + (size_t)tok * 2304 + 1536 + head * 64;
        kv = *(const u128*)(kp + seg * 8);
        vv = *(const u128*)(vp + seg * 8);
      }
      *(u128*)(Kt + kr * 64 + seg * 8) = kv;
      const __bf16* vh = (const __bf16*)&vv;
#pragma unroll
      for (int q = 0; q < 8; ++q) Vt[(seg * 8 + q) * 32 + kr] = vh[q];
    }
    __syncthreads();

    // Scores for two 16-key sub-tiles.
    float p[2][8];
#pragma unroll
    for (int s = 0; s < 2; ++s) {
      int kk = s * 16;
      ABFrag b0, b1;
      b0.u[0] = *(const u128*)(Kt + (kk + r16) * 64 + hiOff);
      b0.u[1] = *(const u128*)(Kt + (kk + r16) * 64 + hiOff + 16);
      b1.u[0] = *(const u128*)(Kt + (kk + r16) * 64 + 32 + hiOff);
      b1.u[1] = *(const u128*)(Kt + (kk + r16) * 64 + 32 + hiOff + 16);
      v8f sc = {};
      sc = WMMA_BF16(aq0.v, b0.v, sc);
      sc = WMMA_BF16(aq1.v, b1.v, sc);
      int lk = kt * 32 + kk + r16;
      bool kvalid = lk < N;
      int kh = 0, kw = 0;
      if (kvalid) {
        if (winPerSide == 0) { kh = lk >> 6; kw = lk & 63; }
        else { kh = lk / size; kw = lk - kh * size; }
      }
#pragma unroll
      for (int i = 0; i < 8; ++i) {
        float v = -1e30f;
        if (kvalid)
          v = sc[i] * scale + RH[wave][(hiOff + i) * 64 + kh]
                            + RW[wave][(hiOff + i) * 64 + kw];
        p[s][i] = v;
      }
    }

    // Online softmax (row reductions across 16-lane halves).
#pragma unroll
    for (int i = 0; i < 8; ++i) {
      float mx = fmaxf(p[0][i], p[1][i]);
      for (int s2 = 1; s2 < 16; s2 <<= 1) mx = fmaxf(mx, __shfl_xor(mx, s2));
      float mnew = fmaxf(mrow[i], mx);
      float alpha = __expf(mrow[i] - mnew);
      p[0][i] = (p[0][i] <= -1e29f) ? 0.f : __expf(p[0][i] - mnew);
      p[1][i] = (p[1][i] <= -1e29f) ? 0.f : __expf(p[1][i] - mnew);
      float rs = p[0][i] + p[1][i];
      for (int s2 = 1; s2 < 16; s2 <<= 1) rs += __shfl_xor(rs, s2);
      lrow[i] = lrow[i] * alpha + rs;
      mrow[i] = mnew;
#pragma unroll
      for (int t = 0; t < 4; ++t) o[t][i] *= alpha;
      Pl[wave][(hiOff + i) * 32 + r16]      = (__bf16)p[0][i];
      Pl[wave][(hiOff + i) * 32 + 16 + r16] = (__bf16)p[1][i];
    }
    __asm__ volatile("s_wait_dscnt 0" ::: "memory");

    // P x V: batch all fragment loads, then 4 back-to-back WMMAs.
    ABFrag pf, vf0, vf1, vf2, vf3;
    pf.u[0]  = *(const u128*)(Pl[wave] + r16 * 32 + hiOff);
    pf.u[1]  = *(const u128*)(Pl[wave] + r16 * 32 + hiOff + 16);
    vf0.u[0] = *(const u128*)(Vt + (r16)      * 32 + hiOff);
    vf0.u[1] = *(const u128*)(Vt + (r16)      * 32 + hiOff + 16);
    vf1.u[0] = *(const u128*)(Vt + (16 + r16) * 32 + hiOff);
    vf1.u[1] = *(const u128*)(Vt + (16 + r16) * 32 + hiOff + 16);
    vf2.u[0] = *(const u128*)(Vt + (32 + r16) * 32 + hiOff);
    vf2.u[1] = *(const u128*)(Vt + (32 + r16) * 32 + hiOff + 16);
    vf3.u[0] = *(const u128*)(Vt + (48 + r16) * 32 + hiOff);
    vf3.u[1] = *(const u128*)(Vt + (48 + r16) * 32 + hiOff + 16);
    o[0] = WMMA_BF16(pf.v, vf0.v, o[0]);
    o[1] = WMMA_BF16(pf.v, vf1.v, o[1]);
    o[2] = WMMA_BF16(pf.v, vf2.v, o[2]);
    o[3] = WMMA_BF16(pf.v, vf3.v, o[3]);
  }

#pragma unroll
  for (int i = 0; i < 8; ++i) {
    if (!rowValid[i]) continue;
    float inv = lrow[i] > 0.f ? 1.f / lrow[i] : 0.f;
#pragma unroll
    for (int t = 0; t < 4; ++t)
      out[(size_t)rowTok[i] * EMBED + head * 64 + t * 16 + r16] =
          (__bf16)(o[t][i] * inv);
  }
}

// ---------------------------------------------------------------------------
// Row LayerNorm.  outB != null -> bf16 row-major; else f32 NCHW-transposed.
// ---------------------------------------------------------------------------
__global__ __launch_bounds__(256) void ln_kernel(
    const float* __restrict__ x, const float* __restrict__ w,
    const float* __restrict__ b, __bf16* __restrict__ outB,
    float* __restrict__ outF, int C, int nchwN)
{
  __shared__ float red[256];
  int row = blockIdx.x, tid = threadIdx.x;
  const float* xr = x + (size_t)row * C;
  float s = 0.f;
  for (int c = tid; c < C; c += 256) s += xr[c];
  red[tid] = s; __syncthreads();
  for (int st = 128; st > 0; st >>= 1) { if (tid < st) red[tid] += red[tid + st]; __syncthreads(); }
  float mu = red[0] / C; __syncthreads();
  float s2 = 0.f;
  for (int c = tid; c < C; c += 256) { float d = xr[c] - mu; s2 += d * d; }
  red[tid] = s2; __syncthreads();
  for (int st = 128; st > 0; st >>= 1) { if (tid < st) red[tid] += red[tid + st]; __syncthreads(); }
  float inv = rsqrtf(red[0] / C + 1e-6f);
  for (int c = tid; c < C; c += 256) {
    float v = (xr[c] - mu) * inv * w[c] + b[c];
    if (outB) outB[(size_t)row * C + c] = (__bf16)v;
    else      outF[(size_t)c * nchwN + row] = v;
  }
}

__global__ void f2b_kernel(const float* __restrict__ x, __bf16* __restrict__ y, size_t n) {
  size_t i = (size_t)blockIdx.x * 256 + threadIdx.x;
  if (i < n) y[i] = (__bf16)x[i];
}

// im2col for 16x16/stride-16 patch conv: (6,1024,1024) -> (4096, 1536) bf16
__global__ void im2col_patch(const float* __restrict__ x, __bf16* __restrict__ A) {
  size_t i = (size_t)blockIdx.x * 256 + threadIdx.x;
  if (i >= (size_t)TOKENS * 1536) return;
  int t = (int)(i / 1536), k = (int)(i % 1536);
  int c = k >> 8, r = (k >> 4) & 15, cc = k & 15;
  int py = t >> 6, px = t & 63;
  A[i] = (__bf16)x[(size_t)c * 1048576 + (size_t)(py * 16 + r) * 1024 + px * 16 + cc];
}

// im2col for 3x3/pad-1 conv over (4096, 256) bf16 -> (4096, 2304) bf16
__global__ void im2col3x3(const __bf16* __restrict__ src, __bf16* __restrict__ A) {
  size_t i = (size_t)blockIdx.x * 256 + threadIdx.x;
  if (i >= (size_t)TOKENS * 2304) return;
  int t = (int)(i / 2304), k = (int)(i % 2304);
  int c = k / 9, kk = k % 9, kh = kk / 3, kw = kk % 3;
  int y = (t >> 6) + kh - 1, x = (t & 63) + kw - 1;
  __bf16 v = (__bf16)0.f;
  if (y >= 0 && y < 64 && x >= 0 && x < 64) v = src[(size_t)(y * 64 + x) * 256 + c];
  A[i] = v;
}

// ---------------------------------------------------------------------------
extern "C" void kernel_launch(void* const* d_in, const int* in_sizes, int n_in,
                              void* d_out, int out_size, void* d_ws, size_t ws_size,
                              hipStream_t stream)
{
  (void)in_sizes; (void)n_in; (void)out_size; (void)ws_size;
  const float* x       = (const float*)d_in[0];
  const float* patch_b = (const float*)d_in[2];
  const float* pos     = (const float*)d_in[3];
  auto blk = [&](int i, int j) { return (const float*)d_in[4 + i * 14 + j]; };
  const float* neck1_w = (const float*)d_in[172];
  const float* ln_a_w  = (const float*)d_in[173];
  const float* ln_a_b  = (const float*)d_in[174];
  const float* neck2_w = (const float*)d_in[175];
  const float* ln_b_w  = (const float*)d_in[176];
  const float* ln_b_b  = (const float*)d_in[177];
  const float* patch_w = (const float*)d_in[1];

  char* ws = (char*)d_ws; size_t off = 0;
  auto carve = [&](size_t bytes) -> void* {
    void* p = ws + off; off = (off + bytes + 255) & ~(size_t)255; return p;
  };
  float*  h     = (float*) carve((size_t)TOKENS * EMBED * 4);
  __bf16* yb    = (__bf16*)carve((size_t)TOKENS * EMBED * 2);
  __bf16* qkvb  = (__bf16*)carve((size_t)TOKENS * 2304 * 2);
  __bf16* attno = (__bf16*)carve((size_t)TOKENS * EMBED * 2);
  __bf16* mlp   = (__bf16*)carve((size_t)TOKENS * MLPDIM * 2);
  __bf16* wb    = (__bf16*)carve((size_t)MLPDIM * EMBED * 2);
  __bf16* pA    = (__bf16*)carve((size_t)TOKENS * 2304 * 2);
  float*  n1    = (float*) carve((size_t)TOKENS * 256 * 4);
  float*  n2    = (float*) carve((size_t)TOKENS * 256 * 4);

  auto conv = [&](const float* s, __bf16* d, size_t n) {
    f2b_kernel<<<(unsigned)((n + 255) / 256), 256, 0, stream>>>(s, d, n);
  };
  auto gemm = [&](const __bf16* A, const __bf16* Wm, const float* bias,
                  const float* add, float* oF, __bf16* oB, int N, int K, int gelu) {
    gemm_bf16_kernel<<<dim3(N / 64, TOKENS / 128), 256, 0, stream>>>(
        A, Wm, bias, add, oF, oB, TOKENS, N, K, gelu);
  };

  // Patch embed (im2col + GEMM, fused bias + pos_embed) -> h (f32)
  im2col_patch<<<(TOKENS * 1536 + 255) / 256, 256, 0, stream>>>(x, pA);
  conv(patch_w, wb, (size_t)EMBED * 1536);
  gemm(pA, wb, patch_b, pos, h, nullptr, EMBED, 1536, 0);

  for (int i = 0; i < 12; ++i) {
    bool isGlobal = (i == 2 || i == 5 || i == 8 || i == 11);
    ln_kernel<<<TOKENS, 256, 0, stream>>>(h, blk(i, 0), blk(i, 1), yb, nullptr, EMBED, 0);
    conv(blk(i, 2), wb, (size_t)2304 * EMBED);
    gemm(yb, wb, blk(i, 3), nullptr, nullptr, qkvb, 2304, EMBED, 0);
    if (isGlobal)
      attn_kernel<<<dim3(HEADS, 64), 128, 0, stream>>>(qkvb, attno, blk(i, 6), blk(i, 7), 4096, 64, 0);
    else
      attn_kernel<<<dim3(25 * HEADS, 4), 128, 0, stream>>>(qkvb, attno, blk(i, 6), blk(i, 7), 196, 14, 5);
    conv(blk(i, 4), wb, (size_t)EMBED * EMBED);
    gemm(attno, wb, blk(i, 5), h, h, nullptr, EMBED, EMBED, 0);   // + residual
    ln_kernel<<<TOKENS, 256, 0, stream>>>(h, blk(i, 8), blk(i, 9), yb, nullptr, EMBED, 0);
    conv(blk(i, 10), wb, (size_t)MLPDIM * EMBED);
    gemm(yb, wb, blk(i, 11), nullptr, nullptr, mlp, MLPDIM, EMBED, 1);  // GELU
    conv(blk(i, 12), wb, (size_t)EMBED * MLPDIM);
    gemm(mlp, wb, blk(i, 13), h, h, nullptr, EMBED, MLPDIM, 0);   // + residual
  }

  // Neck: conv1x1 -> LN(channel) -> conv3x3 -> LN(channel, NCHW f32 out)
  __bf16* hbf = mlp;  // reuse
  conv(h, hbf, (size_t)TOKENS * EMBED);
  conv(neck1_w, wb, (size_t)256 * EMBED);
  gemm(hbf, wb, nullptr, nullptr, n1, nullptr, 256, EMBED, 0);
  ln_kernel<<<TOKENS, 256, 0, stream>>>(n1, ln_a_w, ln_a_b, yb, nullptr, 256, 0);
  im2col3x3<<<(TOKENS * 2304 + 255) / 256, 256, 0, stream>>>(yb, pA);
  conv(neck2_w, wb, (size_t)256 * 2304);
  gemm(pA, wb, nullptr, nullptr, n2, nullptr, 256, 2304, 0);
  ln_kernel<<<TOKENS, 256, 0, stream>>>(n2, ln_b_w, ln_b_b, nullptr, (float*)d_out, 256, TOKENS);
}
